// ModulatedDeformConv2d_35880156791402
// MI455X (gfx1250) — matrix operand order repaired and checked by reading, not compiled, not run on hardware
//
#include <hip/hip_runtime.h>
#include <hip/hip_bf16.h>

typedef __attribute__((ext_vector_type(16))) __bf16        v16bf;
typedef __attribute__((ext_vector_type(8)))  float         v8f;
typedef __attribute__((ext_vector_type(8)))  unsigned int  v8u;

#define HIN   128
#define WIN   128
#define CIN   64
#define OUTC  64
#define BATCH 4
#define P     9
#define KTOT  576           // CIN * P, GEMM K  (k = tap*64 + c)
#define MTILE 128           // pixels per workgroup (one full image row)
#define KCH   32            // k-chunk per WMMA step
#define NKC   18            // KTOT / KCH
#define AROW  40            // A tile row stride in ushorts (80 B, bank-friendly)
#define BROW  40            // B tile row stride in ushorts
#define NOFF  32            // offset-conv N (27 padded to 32)

// workspace layout (bytes)
#define WT_OFF   0                        // bf16 [18][64][32] = 73728 B
#define WOFF_OFF (73728)                  // bf16 [18][32][32] = 36864 B
#define OM_OFF   (110592)                 // f32  [B][32][H*W] = 8388608 B  (py/px/mask)
#define XT_OFF   (110592 + 8388608)       // bf16 [B][H][W][C] = 8388608 B

static __device__ __forceinline__ unsigned short bf16bits(float f) {
    unsigned u = __builtin_bit_cast(unsigned, f);
    unsigned r = u + 0x7FFFu + ((u >> 16) & 1u);      // round-to-nearest-even
    return (unsigned short)(r >> 16);
}

static __device__ __forceinline__ unsigned bcast_bf16(float f) {
    unsigned hb = (unsigned)bf16bits(f);
    return hb | (hb << 16);
}

// packed 2x bf16 fused multiply-add:  d = a*b + c   (V_PK_FMA_BF16, VOP3P)
static __device__ __forceinline__ unsigned pk_fma_bf16(unsigned a, unsigned b,
                                                       unsigned c) {
    unsigned d;
    asm("v_pk_fma_bf16 %0, %1, %2, %3" : "=v"(d) : "v"(a), "v"(b), "v"(c));
    return d;
}

// two aligned 16-byte loads (LDS or global) -> one 32-byte bf16 fragment
static __device__ __forceinline__ v16bf ld_frag(const unsigned short* p,
                                                int i0, int i1) {
    uint4 q0 = *(const uint4*)(p + i0);
    uint4 q1 = *(const uint4*)(p + i1);
    v8u u = {q0.x, q0.y, q0.z, q0.w, q1.x, q1.y, q1.z, q1.w};
    return __builtin_bit_cast(v16bf, u);
}

// 32 bytes of packed bf16 -> 8 dwords
static __device__ __forceinline__ void ld8u(const unsigned short* p,
                                            unsigned* q) {
    uint4 a = *(const uint4*)p;
    uint4 b = *(const uint4*)(p + 8);
    q[0] = a.x; q[1] = a.y; q[2] = a.z; q[3] = a.w;
    q[4] = b.x; q[5] = b.y; q[6] = b.z; q[7] = b.w;
}

// ---------------------------------------------------------------------------
// Kernel 0: both weight tensors -> chunk-major bf16  [kc][n][kk], k = tap*64+c
// ---------------------------------------------------------------------------
__global__ void prep_weights_kernel(const float* __restrict__ weight,
                                    const float* __restrict__ w_off,
                                    unsigned short* __restrict__ wT2,
                                    unsigned short* __restrict__ wOffT) {
    int idx = blockIdx.x * 256 + threadIdx.x;          // 0 .. 55295
    if (idx < KTOT * OUTC) {
        int kc  = idx >> 11;
        int rem = idx & 2047;
        int n   = rem >> 5;
        int kk  = rem & 31;
        int tap = kc >> 1;
        int c   = ((kc & 1) << 5) + kk;
        wT2[idx] = bf16bits(weight[n * KTOT + c * P + tap]);
    } else {
        int j   = idx - KTOT * OUTC;                   // 0 .. 18431
        int kc  = j >> 10;
        int rem = j & 1023;
        int n   = rem >> 5;
        int kk  = rem & 31;
        int tap = kc >> 1;
        int c   = ((kc & 1) << 5) + kk;
        float v = (n < 27) ? w_off[n * KTOT + c * P + tap] : 0.0f;
        wOffT[j] = bf16bits(v);
    }
}

// ---------------------------------------------------------------------------
// Kernel 1: x NCHW f32 -> xT NHWC bf16 (coalesced reads, 8x b128 writes)
// ---------------------------------------------------------------------------
__global__ void xpose_kernel(const float* __restrict__ x,
                             unsigned short* __restrict__ xT) {
    int m  = blockIdx.x * 256 + threadIdx.x;           // pixel 0..65535
    int b  = m >> 14;
    int hw = m & 16383;
    const float* xs = x + ((size_t)b << 20) + hw;
    unsigned q[32];
#pragma unroll
    for (int g = 0; g < 32; ++g) {
        float v0 = xs[(size_t)(2 * g) << 14];
        float v1 = xs[(size_t)(2 * g + 1) << 14];
        q[g] = (unsigned)bf16bits(v0) | ((unsigned)bf16bits(v1) << 16);
    }
    uint4* dst = (uint4*)(xT + ((size_t)m << 6));
#pragma unroll
    for (int i = 0; i < 8; ++i) {
        uint4 t = {q[4 * i], q[4 * i + 1], q[4 * i + 2], q[4 * i + 3]};
        dst[i] = t;
    }
}

// ---------------------------------------------------------------------------
// Kernel 2: offset conv as implicit-GEMM WMMA.  M=128 x N=32, K=576.
// Epilogue applies base grid (ch 0..17) / sigmoid (ch 18..26), so om holds
// ready-to-use py/px/mask.
// ---------------------------------------------------------------------------
__global__ void offset_wmma_kernel(const unsigned short* __restrict__ xT,
                                   const float* __restrict__ b_off,
                                   const unsigned short* __restrict__ wOffT,
                                   float* __restrict__ om) {
    __shared__ __align__(16) unsigned short Atile[MTILE * AROW];
    __shared__ __align__(16) unsigned short Btile[NOFF * BROW];

    const int tid    = threadIdx.x;
    const int waveId = tid >> 5;
    const int lane   = tid & 31;
    const int half   = lane >> 4;
    const int lo16   = lane & 15;

    const int m_base = blockIdx.x * MTILE;
    const int b = m_base >> 14;
    const int h = (m_base >> 7) & (HIN - 1);

    const int mFill = tid >> 1;                        // pixel (== w coord)
    const int cHalf = (tid & 1) << 4;                  // 0 or 16

    v8f acc[2];
#pragma unroll
    for (int t = 0; t < 2; ++t) {
        int o = (t << 4) + lo16;
        float bv = (o < 27) ? b_off[o] : 0.0f;
#pragma unroll
        for (int e = 0; e < 8; ++e) acc[t][e] = bv;
    }

    for (int kc = 0; kc < NKC; ++kc) {
        const int tap = kc >> 1;
        const int ky  = tap / 3, kx = tap % 3;
        const int c0  = (kc & 1) << 5;

        __syncthreads();

        // ---- A tile: shifted window, bf16 copy with zero-padding ----
        {
            int y  = h + ky - 1;
            int xx = mFill + kx - 1;
            bool ok = (y >= 0) & (y < HIN) & (xx >= 0) & (xx < WIN);
            int yc  = min(max(y, 0), HIN - 1);
            int xc_ = min(max(xx, 0), WIN - 1);
            const unsigned short* xp =
                xT + (((size_t)((b << 14) + yc * WIN + xc_)) << 6) + c0 + cHalf;
            uint4 r0 = ((const uint4*)xp)[0];
            uint4 r1 = ((const uint4*)xp)[1];
            uint4 z  = {0u, 0u, 0u, 0u};
            if (!ok) { r0 = z; r1 = z; }
            *(uint4*)&Atile[mFill * AROW + cHalf]     = r0;
            *(uint4*)&Atile[mFill * AROW + cHalf + 8] = r1;
        }

        // ---- B tile: 32n x 32k chunk (128 threads, one uint4 each) ----
        if (tid < 128) {
            const uint4* src = (const uint4*)(wOffT + (kc << 10));
            int n   = tid >> 2;
            int col = (tid & 3) << 3;
            *(uint4*)&Btile[n * BROW + col] = src[tid];
        }

        __syncthreads();

        int rowA = (waveId << 4) + lo16;
        v16bf a = ld_frag(Atile,
                          rowA * AROW + half * 16,
                          rowA * AROW + half * 16 + 8);
#pragma unroll
        for (int t = 0; t < 2; ++t) {
            int nB = (t << 4) + lo16;
            v16bf bf = ld_frag(Btile,
                               nB * BROW + (half << 4),
                               nB * BROW + (half << 4) + 8);
            acc[t] = __builtin_amdgcn_wmma_f32_16x16x32_bf16(
                false, a, false, bf, (short)0, acc[t], false, false);
        }
    }

    // ---- epilogue: fold base grid / sigmoid per channel, then store ----
    const int w0 = (waveId << 4) + (half << 3);        // first of 8 pixels
#pragma unroll
    for (int t = 0; t < 2; ++t) {
        int o = (t << 4) + lo16;
        if (o < 9) {                                   // y-offset -> py
            float by = (float)(h - 1 + o / 3);
#pragma unroll
            for (int e = 0; e < 8; ++e) acc[t][e] += by;
        } else if (o < 18) {                           // x-offset -> px
            int pp = o - 9;
            float bx = (float)(w0 - 1 + pp % 3);
#pragma unroll
            for (int e = 0; e < 8; ++e) acc[t][e] += bx + (float)e;
        } else if (o < 27) {                           // mask -> sigmoid
#pragma unroll
            for (int e = 0; e < 8; ++e)
                acc[t][e] = 1.0f / (1.0f + __expf(-acc[t][e]));
        }
        float* ob = om + ((size_t)b << 19) + ((size_t)o << 14) + (h << 7) + w0;
        float4 s0 = {acc[t][0], acc[t][1], acc[t][2], acc[t][3]};
        float4 s1 = {acc[t][4], acc[t][5], acc[t][6], acc[t][7]};
        *(float4*)(ob)     = s0;
        *(float4*)(ob + 4) = s1;
    }
}

// ---------------------------------------------------------------------------
// Kernel 3: fused bilinear gather (NHWC bf16) + bf16 WMMA GEMM.
// 4-corner blend in packed bf16 via v_pk_fma_bf16 (4 VALU per 2 elements).
// ---------------------------------------------------------------------------
__global__ void deform_wmma_kernel(const unsigned short* __restrict__ xT,
                                   const float* __restrict__ bias,
                                   const float* __restrict__ om,
                                   const unsigned short* __restrict__ wT2,
                                   float* __restrict__ out) {
    __shared__ __align__(16) unsigned short Atile[MTILE * AROW];
    __shared__ __align__(16) unsigned short Btile[OUTC * BROW];

    const int tid    = threadIdx.x;
    const int waveId = tid >> 5;
    const int lane   = tid & 31;
    const int half   = lane >> 4;
    const int lo16   = lane & 15;

    const int m_base = blockIdx.x * MTILE;
    const int b = m_base >> 14;
    const int h = (m_base >> 7) & (HIN - 1);

    const int mFill = tid >> 1;                        // pixel (== w coord)
    const int cHalf = (tid & 1) << 4;                  // 0 or 16

    const int bN   = tid >> 2;
    const int bCol = (tid & 3) << 3;

    v8f acc[4];
#pragma unroll
    for (int t = 0; t < 4; ++t) {
        float bv = bias[(t << 4) + lo16];
#pragma unroll
        for (int e = 0; e < 8; ++e) acc[t][e] = bv;
    }

    const float* omb = om + ((size_t)b << 19);
    const unsigned short* xb = xT + ((size_t)b << 20);  // b * 16384 * 64
    const int pix = (h << 7) + mFill;

    for (int kc = 0; kc < NKC; ++kc) {
        const int p  = kc >> 1;                        // tap
        const int c0 = (kc & 1) << 5;

        __syncthreads();

        // ---- A tile: py/px/mask from om, 4-corner packed-bf16 blend ----
        {
            float pyv = omb[(p << 14) + pix];
            float pxv = omb[((9 + p) << 14) + pix];
            float mk  = omb[((18 + p) << 14) + pix];
            if (kc + 1 < NKC) {
                int pn = (kc + 1) >> 1;
                __builtin_prefetch(&omb[(pn << 14) + pix], 0, 0);
            }
            float y0f = floorf(pyv), x0f = floorf(pxv);
            int iy0 = (int)y0f, ix0 = (int)x0f;
            float wy1 = pyv - y0f, wy0 = 1.0f - wy1;
            float wx1 = pxv - x0f, wx0 = 1.0f - wx1;
            bool vy0 = (iy0 >= 0) & (iy0 < HIN);
            bool vy1 = (iy0 + 1 >= 0) & (iy0 + 1 < HIN);
            bool vx0 = (ix0 >= 0) & (ix0 < WIN);
            bool vx1 = (ix0 + 1 >= 0) & (ix0 + 1 < WIN);
            unsigned pw00 = bcast_bf16(wy0 * wx0 * (float)(vy0 & vx0) * mk);
            unsigned pw01 = bcast_bf16(wy0 * wx1 * (float)(vy0 & vx1) * mk);
            unsigned pw10 = bcast_bf16(wy1 * wx0 * (float)(vy1 & vx0) * mk);
            unsigned pw11 = bcast_bf16(wy1 * wx1 * (float)(vy1 & vx1) * mk);
            int y0c = min(max(iy0, 0), HIN - 1), y1c = min(max(iy0 + 1, 0), HIN - 1);
            int x0c = min(max(ix0, 0), WIN - 1), x1c = min(max(ix0 + 1, 0), WIN - 1);
            int ch = c0 + cHalf;

            unsigned A00[8], A01[8], A10[8], A11[8];
            ld8u(xb + (((size_t)(y0c * WIN + x0c)) << 6) + ch, A00);
            ld8u(xb + (((size_t)(y0c * WIN + x1c)) << 6) + ch, A01);
            ld8u(xb + (((size_t)(y1c * WIN + x0c)) << 6) + ch, A10);
            ld8u(xb + (((size_t)(y1c * WIN + x1c)) << 6) + ch, A11);

            unsigned r[8];
#pragma unroll
            for (int j = 0; j < 8; ++j) {
                unsigned s = pk_fma_bf16(A11[j], pw11, 0u);
                s = pk_fma_bf16(A10[j], pw10, s);
                s = pk_fma_bf16(A01[j], pw01, s);
                r[j] = pk_fma_bf16(A00[j], pw00, s);
            }
            uint4 s0 = {r[0], r[1], r[2], r[3]};
            uint4 s1 = {r[4], r[5], r[6], r[7]};
            *(uint4*)&Atile[mFill * AROW + cHalf]     = s0;
            *(uint4*)&Atile[mFill * AROW + cHalf + 8] = s1;
        }

        // ---- B tile: 64n x 32k chunk, one uint4 per thread ----
        {
            const uint4* src = (const uint4*)(wT2 + (kc << 11));
            *(uint4*)&Btile[bN * BROW + bCol] = src[tid];
        }

        __syncthreads();

        int rowA = (waveId << 4) + lo16;
        v16bf a = ld_frag(Atile,
                          rowA * AROW + half * 16,
                          rowA * AROW + half * 16 + 8);
#pragma unroll
        for (int t = 0; t < 4; ++t) {
            int nB = (t << 4) + lo16;
            v16bf bf = ld_frag(Btile,
                               nB * BROW + (half << 4),
                               nB * BROW + (half << 4) + 8);
            acc[t] = __builtin_amdgcn_wmma_f32_16x16x32_bf16(
                false, a, false, bf, (short)0, acc[t], false, false);
        }
    }

#pragma unroll
    for (int t = 0; t < 4; ++t) {
        int o = (t << 4) + lo16;
        float* ob = out + ((size_t)(b * OUTC + o) << 14) + (h << 7)
                  + (waveId << 4) + (half << 3);
        float4 s0 = {acc[t][0], acc[t][1], acc[t][2], acc[t][3]};
        float4 s1 = {acc[t][4], acc[t][5], acc[t][6], acc[t][7]};
        *(float4*)(ob)     = s0;
        *(float4*)(ob + 4) = s1;
    }
}

// ---------------------------------------------------------------------------
extern "C" void kernel_launch(void* const* d_in, const int* in_sizes, int n_in,
                              void* d_out, int out_size, void* d_ws, size_t ws_size,
                              hipStream_t stream) {
    const float* x      = (const float*)d_in[0];
    const float* w_off  = (const float*)d_in[1];
    const float* b_off  = (const float*)d_in[2];
    const float* weight = (const float*)d_in[3];
    const float* bias   = (const float*)d_in[4];
    float* out = (float*)d_out;

    char* ws = (char*)d_ws;
    unsigned short* wT2   = (unsigned short*)(ws + WT_OFF);
    unsigned short* wOffT = (unsigned short*)(ws + WOFF_OFF);
    float*          om    = (float*)(ws + OM_OFF);
    unsigned short* xT    = (unsigned short*)(ws + XT_OFF);

    prep_weights_kernel<<<(KTOT * OUTC + KTOT * NOFF) / 256, 256, 0, stream>>>(
        weight, w_off, wT2, wOffT);
    xpose_kernel<<<(BATCH * HIN * WIN) / 256, 256, 0, stream>>>(x, xT);
    offset_wmma_kernel<<<(BATCH * HIN * WIN) / MTILE, 256, 0, stream>>>(
        xT, b_off, wOffT, om);
    deform_wmma_kernel<<<(BATCH * HIN * WIN) / MTILE, 256, 0, stream>>>(
        xT, bias, om, wT2, out);
}
